// GSATSubgraph_44590350467894
// MI455X (gfx1250) — matrix operand order repaired and checked by reading, not compiled, not run on hardware
//
#include <hip/hip_runtime.h>
#include <cstdint>
#include <cstddef>

typedef __attribute__((ext_vector_type(16))) _Float16 v16h;
typedef __attribute__((ext_vector_type(8)))  _Float16 v8h;
typedef __attribute__((ext_vector_type(8)))  float    v8f;

// ---------------------------------------------------------------------------
// Problem constants
// ---------------------------------------------------------------------------
constexpr int Bc = 64;
constexpr int Cc = 384;
constexpr int Hh = 28, Ww = 28;
constexpr int Nn = Hh * Ww;            // 784
constexpr int Kk = 9;
constexpr int NN = Bc * Nn;            // 50176 rows (batch folded into M)
constexpr int C2 = 2 * Cc;             // 768
constexpr int C4 = 4 * Cc;             // 1536
constexpr int ME = NN * Kk;            // 451584 edges

// ---------------------------------------------------------------------------
// WMMA helpers (wave32; C/D layout per CDNA5 ISA §7.12.2; A/B use this file's contiguous K order)
//
// A 16x32 f16: lane (m = lane&15, hi = lane>>4) element e holds K = e + 16*hi
//   -> two contiguous 16B loads from a row-major row.
// B 32x16 f16: lane (n = lane&15, hi) element e holds K = e + 16*hi, column n
//   -> 16 contiguous f16 from a COLUMN-MAJOR operand (K contiguous per n)
//   -> two contiguous 16B loads. All B operands are stored column-major.
// C/D 16x16 f32: element i -> row i + 8*hi, col lane&15.
// ---------------------------------------------------------------------------
__device__ inline v16h ld_frag16(const _Float16* p0, const _Float16* p1) {
  v8h lo = *(const v8h*)p0;
  v8h hi = *(const v8h*)p1;
  return __builtin_shufflevector(lo, hi, 0, 1, 2, 3, 4, 5, 6, 7,
                                 8, 9, 10, 11, 12, 13, 14, 15);
}
__device__ inline v16h ld_frag16_lds(const _Float16* p0, const _Float16* p1) {
  v8h lo = *(const v8h*)p0;   // ds_load_b128
  v8h hi = *(const v8h*)p1;
  return __builtin_shufflevector(lo, hi, 0, 1, 2, 3, 4, 5, 6, 7,
                                 8, 9, 10, 11, 12, 13, 14, 15);
}
__device__ inline v8f wmma_f16(v16h a, v16h b, v8f c) {
  return __builtin_amdgcn_wmma_f32_16x16x32_f16(false, a, false, b, (short)0, c,
                                                false, false);
}

// ---------------------------------------------------------------------------
// Weight conversion: f32 -> f16, direct or transposed-to-column-major [N][K]
// ---------------------------------------------------------------------------
__global__ void k_cvt_direct(const float* __restrict__ s,
                             _Float16* __restrict__ d, int n) {
  int i = blockIdx.x * blockDim.x + threadIdx.x;
  if (i < n) d[i] = (_Float16)s[i];
}
// src row-major [K,N] f32 -> dst [N][K] f16 (column-major B operand)
__global__ void k_cvt_T(const float* __restrict__ src,
                        _Float16* __restrict__ dst, int K, int N) {
  int i = blockIdx.x * blockDim.x + threadIdx.x;
  if (i >= K * N) return;
  int k = i / N, n = i - k * N;
  dst[(size_t)n * K + k] = (_Float16)src[i];
}

// x [B,C,N] f32 -> xT [B*N, C] f16
__global__ void k_prep_x(const float* __restrict__ x, _Float16* __restrict__ xT) {
  int node = blockIdx.x;               // b*Nn + n
  int b = node / Nn, n = node - b * Nn;
  for (int c = threadIdx.x; c < Cc; c += blockDim.x)
    xT[(size_t)node * Cc + c] = (_Float16)x[((size_t)b * Cc + c) * Nn + n];
}

// ---------------------------------------------------------------------------
// WMMA GEMM (LDS-free): out = epilogue(A[MxK] @ B[KxN] + bias)
//   Wt is the B operand stored column-major: Wt[n*K + k].
//   Each wave computes a 16x64 strip: 4 accumulators share one A fragment.
//   mode: 0=bias, 1=bias+relu, 2=bias+BN, 3=bias+BN+gelu, 4=bias+BN+residual
// ---------------------------------------------------------------------------
__global__ __launch_bounds__(32)
void k_gemm_f16(const _Float16* __restrict__ A, int lda,
                const _Float16* __restrict__ Wt,
                const float* __restrict__ bias,
                float* __restrict__ outF, _Float16* __restrict__ outH,
                int M, int N, int K, int mode,
                const float* __restrict__ bn_g, const float* __restrict__ bn_b,
                const float* __restrict__ bn_m, const float* __restrict__ bn_v,
                const float* __restrict__ resid) {
  int lane = threadIdx.x;
  int nl = lane & 15, hi = lane >> 4;
  int m0 = blockIdx.y * 16;
  int n0 = blockIdx.x * 64;

  const _Float16* arow = A + (size_t)(m0 + nl) * lda;
  const _Float16* bc[4];
#pragma unroll
  for (int t = 0; t < 4; ++t) bc[t] = Wt + (size_t)(n0 + t * 16 + nl) * K;

  v8f acc[4];
#pragma unroll
  for (int t = 0; t < 4; ++t) acc[t] = (v8f){0.f,0.f,0.f,0.f,0.f,0.f,0.f,0.f};

  for (int k0 = 0; k0 < K; k0 += 32) {
    const _Float16* ap = arow + k0 + 16 * hi;
    v16h a = ld_frag16(ap, ap + 8);
#pragma unroll
    for (int t = 0; t < 4; ++t) {
      const _Float16* bp = bc[t] + k0 + 16 * hi;
      v16h b = ld_frag16(bp, bp + 8);
      acc[t] = wmma_f16(a, b, acc[t]);
    }
  }

  // Epilogue: acc[t] element i -> (row m0+i+8*hi, col n0+t*16+nl)
#pragma unroll
  for (int t = 0; t < 4; ++t) {
    int n = n0 + t * 16 + nl;
    float bs = bias ? bias[n] : 0.f;
    float sc = 1.f, sh = 0.f;
    if (mode >= 2) {
      sc = bn_g[n] * rsqrtf(bn_v[n] + 1e-5f);
      sh = bn_b[n] - bn_m[n] * sc;
    }
#pragma unroll
    for (int i = 0; i < 8; ++i) {
      int m = m0 + i + 8 * hi;
      float v = acc[t][i] + bs;
      if (mode == 1) v = fmaxf(v, 0.f);
      if (mode >= 2) v = v * sc + sh;
      if (mode == 3) {                 // gelu (tanh approximation)
        float u = 0.7978845608f * (v + 0.044715f * v * v * v);
        v = 0.5f * v * (1.f + tanhf(u));
      }
      size_t o = (size_t)m * N + n;
      if (mode == 4) v += resid[o];
      if (outF) outF[o] = v;
      if (outH) outH[o] = (_Float16)v;
    }
  }
}

// ---------------------------------------------------------------------------
// L2 normalize rows of h [NN x C]; writes nf back in place + f16 copy
// ---------------------------------------------------------------------------
__global__ __launch_bounds__(128)
void k_l2norm(float* __restrict__ h, _Float16* __restrict__ nfh) {
  __shared__ float red[128];
  int node = blockIdx.x;
  float s = 0.f;
  for (int c = threadIdx.x; c < Cc; c += 128) {
    float v = h[(size_t)node * Cc + c];
    s += v * v;
  }
  red[threadIdx.x] = s;
  __syncthreads();
  for (int st = 64; st > 0; st >>= 1) {
    if (threadIdx.x < st) red[threadIdx.x] += red[threadIdx.x + st];
    __syncthreads();
  }
  float inv = 1.f / fmaxf(sqrtf(red[0]), 1e-12f);
  for (int c = threadIdx.x; c < Cc; c += 128) {
    float v = h[(size_t)node * Cc + c] * inv;
    h[(size_t)node * Cc + c] = v;
    nfh[(size_t)node * Cc + c] = (_Float16)v;
  }
}

// ---------------------------------------------------------------------------
// Fused similarity + top-9.  sim = nf @ nf^T per batch; the B operand column
// IS a contiguous nf row, so B fragments load directly (no LDS, no staging).
// Per (batch, 16-row tile): 49 column tiles x 12 K-steps of WMMA; per-row
// sorted top-9 with self-exclusion, on the 16x16 tile dumped to LDS.
// ---------------------------------------------------------------------------
__global__ __launch_bounds__(32)
void k_sim_topk(const _Float16* __restrict__ nf, int* __restrict__ nbidx) {
  __shared__ float Tile[16 * 16];
  int b = blockIdx.y;
  int rt = blockIdx.x;                 // 0..48
  const _Float16* base = nf + (size_t)b * Nn * Cc;
  int lane = threadIdx.x;
  int nl = lane & 15, hi = lane >> 4;
  int row = rt * 16 + nl;
  const _Float16* arow = base + (size_t)row * Cc;

  float tv[9]; int ti[9];
#pragma unroll
  for (int j = 0; j < 9; ++j) { tv[j] = -1e30f; ti[j] = 0; }

  for (int ct = 0; ct < 49; ++ct) {
    const _Float16* bcol = base + (size_t)(ct * 16 + nl) * Cc;
    v8f acc = {0.f, 0.f, 0.f, 0.f, 0.f, 0.f, 0.f, 0.f};
#pragma unroll 4
    for (int ks = 0; ks < 12; ++ks) {
      int k0 = ks * 32;
      const _Float16* ap = arow + k0 + 16 * hi;
      const _Float16* bp = bcol + k0 + 16 * hi;
      v16h a = ld_frag16(ap, ap + 8);
      v16h bb = ld_frag16(bp, bp + 8);
      acc = wmma_f16(a, bb, acc);
    }
#pragma unroll
    for (int i = 0; i < 8; ++i) Tile[(i + 8 * hi) * 16 + nl] = acc[i];
    __syncthreads();
    if (lane < 16) {                   // lanes 0..15 own one row each
#pragma unroll
      for (int cc = 0; cc < 16; ++cc) {
        int gcol = ct * 16 + cc;
        if (gcol == row) continue;     // self-loop exclusion (-4*eye)
        float v = Tile[lane * 16 + cc];
        if (v > tv[8]) {
          int p = 8;
          while (p > 0 && v > tv[p - 1]) { tv[p] = tv[p-1]; ti[p] = ti[p-1]; --p; }
          tv[p] = v; ti[p] = gcol;
        }
      }
    }
    __syncthreads();
  }
  if (lane < 16) {
    int* o = nbidx + ((size_t)b * Nn + row) * Kk;
#pragma unroll
    for (int j = 0; j < 9; ++j) o[j] = ti[j];
  }
}

// ---------------------------------------------------------------------------
// GIN aggregate: out = (1+eps)*nf[node] + sum_j w_j * nf[neighbor_j]  (f16)
// ---------------------------------------------------------------------------
__global__ __launch_bounds__(128)
void k_gather_agg(const float* __restrict__ nf, const int* __restrict__ nbidx,
                  const float* __restrict__ w, const float* __restrict__ eps,
                  _Float16* __restrict__ out) {
  int node = blockIdx.x;
  int b = node / Nn;
  const int* idx = nbidx + (size_t)node * Kk;
  const float* wr = w ? w + (size_t)node * Kk : nullptr;
  float e = 1.f + eps[0];
  for (int c = threadIdx.x; c < Cc; c += 128) {
    float s = e * nf[(size_t)node * Cc + c];
#pragma unroll
    for (int j = 0; j < 9; ++j) {
      float wv = wr ? wr[j] : 1.f;
      s += wv * nf[((size_t)b * Nn + idx[j]) * Cc + c];
    }
    out[(size_t)node * Cc + c] = (_Float16)s;
  }
}

// ---------------------------------------------------------------------------
// Fused edge attention: per block, 16 edges. A row = concat(emb[dst],emb[src])
// gathered to LDS with 128-bit copies; eh = relu(A @ att_w1 + b1) evaluated
// tile-by-tile (4 column tiles per A fragment), dotted with att_w2 on the fly,
// reduced across lanes, + b2, sigmoid -> edge_att.
// w1t: att_w1 column-major [384][768].
// ---------------------------------------------------------------------------
__global__ __launch_bounds__(32)
void k_edge_att(const _Float16* __restrict__ emb, const int* __restrict__ nbidx,
                const _Float16* __restrict__ w1t,
                const float* __restrict__ b1, const float* __restrict__ w2,
                const float* __restrict__ b2, float* __restrict__ eatt) {
  __shared__ _Float16 At[16 * C2];     // 24 KB, row-major per edge
  __shared__ float red[16 * 16];
  __shared__ int dstrow[16], srcrow[16];
  int e0 = blockIdx.x * 16;
  int lane = threadIdx.x;
  int nl = lane & 15, hi = lane >> 4;

  if (lane < 16) {
    int e = e0 + lane;
    int node = e / Kk, j = e - node * Kk;
    int b = node / Nn;
    dstrow[lane] = node;
    srcrow[lane] = b * Nn + nbidx[(size_t)node * Kk + j];
  }
  __syncthreads();
  // Gather A: 16 edges x 768 f16, 128-bit granules
  for (int t = lane; t < 16 * 96; t += 32) {
    int er = t / 96, q0 = (t - er * 96) * 8;
    const _Float16* src = (q0 < Cc)
        ? emb + (size_t)dstrow[er] * Cc + q0
        : emb + (size_t)srcrow[er] * Cc + (q0 - Cc);
    *(v8h*)(At + er * C2 + q0) = *(const v8h*)src;
  }
  __syncthreads();

  float part[8];
#pragma unroll
  for (int i = 0; i < 8; ++i) part[i] = 0.f;
  const _Float16* arow = At + nl * C2;

  for (int cq = 0; cq < 6; ++cq) {     // 6 x (4 x 16) = 384 output columns
    int n0 = cq * 64;
    const _Float16* bc[4];
#pragma unroll
    for (int t = 0; t < 4; ++t) bc[t] = w1t + (size_t)(n0 + t * 16 + nl) * C2;
    v8f acc[4];
#pragma unroll
    for (int t = 0; t < 4; ++t) acc[t] = (v8f){0.f,0.f,0.f,0.f,0.f,0.f,0.f,0.f};
    for (int k0 = 0; k0 < C2; k0 += 32) {
      const _Float16* ap = arow + k0 + 16 * hi;
      v16h a = ld_frag16_lds(ap, ap + 8);
#pragma unroll
      for (int t = 0; t < 4; ++t) {
        const _Float16* bp = bc[t] + k0 + 16 * hi;
        v16h bb = ld_frag16(bp, bp + 8);
        acc[t] = wmma_f16(a, bb, acc[t]);
      }
    }
#pragma unroll
    for (int t = 0; t < 4; ++t) {
      int n = n0 + t * 16 + nl;
      float bv = b1[n], wv = w2[n];
#pragma unroll
      for (int i = 0; i < 8; ++i) part[i] += fmaxf(acc[t][i] + bv, 0.f) * wv;
    }
  }
#pragma unroll
  for (int i = 0; i < 8; ++i) red[(i + 8 * hi) * 16 + nl] = part[i];
  __syncthreads();
  if (lane < 16) {
    float s = 0.f;
#pragma unroll
    for (int c = 0; c < 16; ++c) s += red[lane * 16 + c];
    s += b2[0];
    eatt[e0 + lane] = 1.f / (1.f + __expf(-s));
  }
}

// y [B*N, C] f32 -> node_embeddings [B,C,H,W]
__global__ void k_final_out(const float* __restrict__ y, float* __restrict__ out) {
  size_t i = (size_t)blockIdx.x * blockDim.x + threadIdx.x;
  if (i >= (size_t)Bc * Cc * Nn) return;
  int n = (int)(i % Nn);
  size_t r = i / Nn;
  int c = (int)(r % Cc);
  int b = (int)(r / Cc);
  out[i] = y[((size_t)b * Nn + n) * Cc + c];
}

// ---------------------------------------------------------------------------
// Host orchestration
// ---------------------------------------------------------------------------
extern "C" void kernel_launch(void* const* d_in, const int* in_sizes, int n_in,
                              void* d_out, int out_size, void* d_ws, size_t ws_size,
                              hipStream_t stream) {
  (void)in_sizes; (void)n_in; (void)out_size; (void)ws_size;

  const float* x      = (const float*)d_in[0];
  const float* fc1_w  = (const float*)d_in[2];
  const float* fc1_b  = (const float*)d_in[3];
  const float* bn1_g  = (const float*)d_in[4];
  const float* bn1_b  = (const float*)d_in[5];
  const float* bn1_m  = (const float*)d_in[6];
  const float* bn1_v  = (const float*)d_in[7];
  const float* gin_eps= (const float*)d_in[8];
  const float* gin_w1 = (const float*)d_in[9];
  const float* gin_b1 = (const float*)d_in[10];
  const float* gin_w2 = (const float*)d_in[11];
  const float* gin_b2 = (const float*)d_in[12];
  const float* att_w1 = (const float*)d_in[13];
  const float* att_b1 = (const float*)d_in[14];
  const float* att_w2 = (const float*)d_in[15];
  const float* att_b2 = (const float*)d_in[16];
  const float* ffn_w1 = (const float*)d_in[17];
  const float* ffn_b1 = (const float*)d_in[18];
  const float* fbn1_g = (const float*)d_in[19];
  const float* fbn1_b = (const float*)d_in[20];
  const float* fbn1_m = (const float*)d_in[21];
  const float* fbn1_v = (const float*)d_in[22];
  const float* ffn_w2 = (const float*)d_in[23];
  const float* ffn_b2 = (const float*)d_in[24];
  const float* fbn2_g = (const float*)d_in[25];
  const float* fbn2_b = (const float*)d_in[26];
  const float* fbn2_m = (const float*)d_in[27];
  const float* fbn2_v = (const float*)d_in[28];

  float* out_edge = (float*)d_out;                 // [ME]
  float* out_node = (float*)d_out + ME;            // [B*C*N]

  // Workspace bump allocator (256B aligned)
  char* ws = (char*)d_ws;
  size_t off = 0;
  auto alloc = [&](size_t bytes) {
    size_t o = off;
    off = (off + bytes + 255) & ~(size_t)255;
    return (void*)(ws + o);
  };
  _Float16* xT_h    = (_Float16*)alloc((size_t)NN * Cc * 2);
  float*    hbuf    = (float*)   alloc((size_t)NN * Cc * 4);  // h -> nf -> y
  _Float16* nf_h    = (_Float16*)alloc((size_t)NN * Cc * 2);
  int*      nbidx   = (int*)     alloc((size_t)NN * Kk * 4);
  _Float16* hagg_h  = (_Float16*)alloc((size_t)NN * Cc * 2);
  _Float16* mid_h   = (_Float16*)alloc((size_t)NN * C2 * 2);
  _Float16* emb_h   = (_Float16*)alloc((size_t)NN * Cc * 2);
  float*    ne_f    = (float*)   alloc((size_t)NN * Cc * 4);
  _Float16* ne_h    = (_Float16*)alloc((size_t)NN * Cc * 2);
  _Float16* fmid_h  = (_Float16*)alloc((size_t)NN * C4 * 2);
  // f16 B operands, all column-major [N][K]
  _Float16* wfc1_t  = (_Float16*)alloc((size_t)Cc * Cc * 2);  // [C][C]
  _Float16* wg1_t   = (_Float16*)alloc((size_t)C2 * Cc * 2);  // [C2][C]
  _Float16* wg2_t   = (_Float16*)alloc((size_t)Cc * C2 * 2);  // [C][C2]
  _Float16* wa1_t   = (_Float16*)alloc((size_t)Cc * C2 * 2);  // [C][C2]
  _Float16* wf1_t   = (_Float16*)alloc((size_t)C4 * Cc * 2);  // [C4][C]
  _Float16* wf2_t   = (_Float16*)alloc((size_t)Cc * C4 * 2);  // [C][C4]

  // fc1 needs B[k][n] = fc1_w[n][k] -> column-major B == fc1_w as-is
  k_cvt_direct<<<(Cc * Cc + 255) / 256, 256, 0, stream>>>(fc1_w, wfc1_t, Cc * Cc);
  auto cvtT = [&](const float* s, _Float16* d, int K, int N) {
    k_cvt_T<<<(K * N + 255) / 256, 256, 0, stream>>>(s, d, K, N);
  };
  cvtT(gin_w1, wg1_t, Cc, C2);
  cvtT(gin_w2, wg2_t, C2, Cc);
  cvtT(att_w1, wa1_t, C2, Cc);
  cvtT(ffn_w1, wf1_t, Cc, C4);
  cvtT(ffn_w2, wf2_t, C4, Cc);

  // 1) x -> [B*N, C] f16
  k_prep_x<<<NN, 128, 0, stream>>>(x, xT_h);

  // 2) fc1 + BN
  k_gemm_f16<<<dim3(Cc / 64, NN / 16), 32, 0, stream>>>(
      xT_h, Cc, wfc1_t, fc1_b, hbuf, nullptr, NN, Cc, Cc, 2,
      bn1_g, bn1_b, bn1_m, bn1_v, nullptr);

  // 3) L2 normalize -> nf (f32 in place + f16)
  k_l2norm<<<NN, 128, 0, stream>>>(hbuf, nf_h);

  // 4) similarity + top-9 neighbors
  k_sim_topk<<<dim3(Nn / 16, Bc), 32, 0, stream>>>(nf_h, nbidx);

  // 5) GIN pass 1 (unit weights)
  k_gather_agg<<<NN, 128, 0, stream>>>(hbuf, nbidx, nullptr, gin_eps, hagg_h);
  k_gemm_f16<<<dim3(C2 / 64, NN / 16), 32, 0, stream>>>(
      hagg_h, Cc, wg1_t, gin_b1, nullptr, mid_h, NN, C2, Cc, 1,
      nullptr, nullptr, nullptr, nullptr, nullptr);
  k_gemm_f16<<<dim3(Cc / 64, NN / 16), 32, 0, stream>>>(
      mid_h, C2, wg2_t, gin_b2, nullptr, emb_h, NN, Cc, C2, 0,
      nullptr, nullptr, nullptr, nullptr, nullptr);

  // 6) edge attention -> d_out[0:ME]
  k_edge_att<<<ME / 16, 32, 0, stream>>>(emb_h, nbidx, wa1_t, att_b1, att_w2,
                                         att_b2, out_edge);

  // 7) GIN pass 2 (edge-attention weights)
  k_gather_agg<<<NN, 128, 0, stream>>>(hbuf, nbidx, out_edge, gin_eps, hagg_h);
  k_gemm_f16<<<dim3(C2 / 64, NN / 16), 32, 0, stream>>>(
      hagg_h, Cc, wg1_t, gin_b1, nullptr, mid_h, NN, C2, Cc, 1,
      nullptr, nullptr, nullptr, nullptr, nullptr);
  k_gemm_f16<<<dim3(Cc / 64, NN / 16), 32, 0, stream>>>(
      mid_h, C2, wg2_t, gin_b2, ne_f, ne_h, NN, Cc, C2, 0,
      nullptr, nullptr, nullptr, nullptr, nullptr);

  // 8) FFN: BN+gelu, then BN + residual(ne); y overwrites hbuf
  k_gemm_f16<<<dim3(C4 / 64, NN / 16), 32, 0, stream>>>(
      ne_h, Cc, wf1_t, ffn_b1, nullptr, fmid_h, NN, C4, Cc, 3,
      fbn1_g, fbn1_b, fbn1_m, fbn1_v, nullptr);
  k_gemm_f16<<<dim3(Cc / 64, NN / 16), 32, 0, stream>>>(
      fmid_h, C4, wf2_t, ffn_b2, hbuf, nullptr, NN, Cc, C4, 4,
      fbn2_g, fbn2_b, fbn2_m, fbn2_v, ne_f);

  // 9) [B*N,C] -> [B,C,H,W]
  size_t tot = (size_t)Bc * Cc * Nn;
  k_final_out<<<(unsigned)((tot + 255) / 256), 256, 0, stream>>>(hbuf, out_node);
}
